// Attention_15281493639657
// MI455X (gfx1250) — matrix operand order repaired and checked by reading, not compiled, not run on hardware
//
#include <hip/hip_runtime.h>

// ---------------------------------------------------------------------------
// Multi-head attention for MI455X (gfx1250), wave32 + WMMA f16->f32.
//   x:[S,B,D]=[2048,2,1024], H=16 heads, DH=64, scale=DH^-0.5=0.125
// Pipeline: f32->f16 casts -> QKV WMMA GEMMs (32x32 register-blocked, +bias,
//           V stored transposed) -> flash-attention (online softmax, LDS
//           C->A relayout) -> output WMMA GEMM (32x32 blocked, +bias, f32).
// ---------------------------------------------------------------------------

typedef __attribute__((ext_vector_type(16))) _Float16 v16h;
typedef __attribute__((ext_vector_type(8)))  _Float16 v8h;
typedef __attribute__((ext_vector_type(8)))  float    v8f;

#define S_LEN   2048
#define BATCH   2
#define DMODEL  1024
#define NHEAD   16
#define DHEAD   64
#define NROWS   (S_LEN * BATCH)   /* 4096 */
#define SCALEF  0.125f

union V16U { v16h v; v8h h[2]; };

// WMMA wrapper: D = A(16x32 f16) * B(32x16 f16) + C(16x16 f32)
__device__ __forceinline__ v8f wmma32(v16h a, v16h b, v8f c) {
  return __builtin_amdgcn_wmma_f32_16x16x32_f16(
      /*neg_a=*/false, a, /*neg_b=*/false, b,
      /*c_mod=*/(short)0, c, /*reuse_a=*/false, /*reuse_b=*/false);
}

// A-fragment (16x32, row-major source, leading dim ld):
//   lane&15 selects row M; halves [0..7] hold K = rg*8+i, [8..15] hold K = 16+rg*8+i
__device__ __forceinline__ v16h load_a_f16(const _Float16* __restrict__ p,
                                           int ld, int row0, int col0) {
  const int lane = threadIdx.x & 31;
  const int rl = lane & 15, rg = lane >> 4;
  const _Float16* q = p + (size_t)(row0 + rl) * ld + col0;
  V16U u;
  u.h[0] = *(const v8h*)(q + rg * 8);
  u.h[1] = *(const v8h*)(q + 16 + rg * 8);
  return u.v;
}

// B-fragment (32x16). Source is "B transposed" row-major: mem[n][k], ld = k-stride.
//   lane&15 selects column N; halves [0..7] hold K = rg*8+i, [8..15] hold K = 16+rg*8+i
__device__ __forceinline__ v16h load_b_f16(const _Float16* __restrict__ p,
                                           int ld, int n0, int k0) {
  const int lane = threadIdx.x & 31;
  const int rl = lane & 15, rg = lane >> 4;
  const _Float16* q = p + (size_t)(n0 + rl) * ld + k0;
  V16U u;
  u.h[0] = *(const v8h*)(q + rg * 8);
  u.h[1] = *(const v8h*)(q + 16 + rg * 8);
  return u.v;
}

// ---------------------------------------------------------------------------
__global__ void cast_f32_to_f16(const float* __restrict__ in,
                                _Float16* __restrict__ out, int n) {
  for (int i = blockIdx.x * blockDim.x + threadIdx.x; i < n;
       i += gridDim.x * blockDim.x)
    out[i] = (_Float16)in[i];
}

// ---------------------------------------------------------------------------
// Scatter-store one 16x16 f32 accumulator tile of Q/K/V with bias.
// mode 0/1: [b][h][s][dh] row-major ; mode 2: [b][h][dh][s] (V transposed)
__device__ __forceinline__ void store_qkv_tile(v8f acc, int nbase, int o,
                                               float bi, _Float16* __restrict__ dst,
                                               int mode) {
  const int h = o >> 6, dh = o & 63;
#pragma unroll
  for (int r = 0; r < 8; ++r) {
    const int n = nbase + r;             // n = s*BATCH + b
    const int s = n >> 1, bb = n & 1;
    const float val = acc[r] + bi;
    if (mode == 2)
      dst[(((size_t)bb * NHEAD + h) * DHEAD + dh) * S_LEN + s] = (_Float16)val;
    else
      dst[(((size_t)bb * NHEAD + h) * S_LEN + s) * DHEAD + dh] = (_Float16)val;
  }
}

// QKV projection: dst = X[4096x1024] @ W^T + bias, W row-major [O=1024, D=1024].
// 32x32 register blocking: 4 WMMAs per 4 fragment loads (2A + 2B) per k-step.
__global__ void qkv_gemm(const _Float16* __restrict__ X,
                         const _Float16* __restrict__ W,
                         const float* __restrict__ bias,
                         _Float16* __restrict__ dst, int mode) {
  const int wave = blockIdx.x * (blockDim.x >> 5) + (threadIdx.x >> 5);
  const int tn = wave % (DMODEL / 32);   // output-feature 32-tile
  const int tm = wave / (DMODEL / 32);   // row 32-tile
  const int lane = threadIdx.x & 31;
  const int rl = lane & 15, rg = lane >> 4;
  const int m0 = tm * 32, n0 = tn * 32;

  v8f c00 = {}, c01 = {}, c10 = {}, c11 = {};
  for (int k0 = 0; k0 < DMODEL; k0 += 32) {
    const v16h a0 = load_a_f16(X, DMODEL, m0,      k0);
    const v16h a1 = load_a_f16(X, DMODEL, m0 + 16, k0);
    const v16h b0 = load_b_f16(W, DMODEL, n0,      k0);
    const v16h b1 = load_b_f16(W, DMODEL, n0 + 16, k0);
    c00 = wmma32(a0, b0, c00);
    c01 = wmma32(a0, b1, c01);
    c10 = wmma32(a1, b0, c10);
    c11 = wmma32(a1, b1, c11);
  }

  const int o0 = n0 + rl, o1 = o0 + 16;
  const float bi0 = bias[o0], bi1 = bias[o1];
  const int nb0 = m0 + rg * 8, nb1 = nb0 + 16;
  store_qkv_tile(c00, nb0, o0, bi0, dst, mode);
  store_qkv_tile(c01, nb0, o1, bi1, dst, mode);
  store_qkv_tile(c10, nb1, o0, bi0, dst, mode);
  store_qkv_tile(c11, nb1, o1, bi1, dst, mode);
}

// ---------------------------------------------------------------------------
// Flash attention: one wave per (b*H+h, 16-query tile). Qh/Kh: [bh][S][64],
// Vt: [bh][64][S]. Output Oh: [n=s*B+b][h*64+dh] f16.
__global__ void flash_attn(const _Float16* __restrict__ Qh,
                           const _Float16* __restrict__ Kh,
                           const _Float16* __restrict__ Vt,
                           _Float16* __restrict__ Oh) {
  __shared__ __align__(32) _Float16 Pb[4][16 * 32];
  const int wv = threadIdx.x >> 5;
  _Float16* lds = Pb[wv];

  const int gw = blockIdx.x * 4 + wv;
  const int st = gw % (S_LEN / 16);
  const int bh = gw / (S_LEN / 16);          // b*NHEAD + h
  const int s0 = st * 16;
  const int lane = threadIdx.x & 31;
  const int rl = lane & 15, rg = lane >> 4;

  const _Float16* Qb = Qh + ((size_t)bh * S_LEN + s0) * DHEAD;
  const _Float16* Kb = Kh + (size_t)bh * S_LEN * DHEAD;
  const _Float16* Vb = Vt + (size_t)bh * DHEAD * S_LEN;

  const v16h aq0 = load_a_f16(Qb, DHEAD, 0, 0);
  const v16h aq1 = load_a_f16(Qb, DHEAD, 0, 32);

  v8f acc0 = {}, acc1 = {}, acc2 = {}, acc3 = {};
  float m[8], l[8];
#pragma unroll
  for (int r = 0; r < 8; ++r) { m[r] = -1e30f; l[r] = 0.f; }

  for (int t0 = 0; t0 < S_LEN; t0 += 32) {
    // ---- scores: two 16-key tiles, K-dim = DHEAD = 64 (2 wmma each) ----
    v8f sc0 = {}, sc1 = {};
    sc0 = wmma32(aq0, load_b_f16(Kb, DHEAD, t0,       0),  sc0);
    sc0 = wmma32(aq1, load_b_f16(Kb, DHEAD, t0,       32), sc0);
    sc1 = wmma32(aq0, load_b_f16(Kb, DHEAD, t0 + 16,  0),  sc1);
    sc1 = wmma32(aq1, load_b_f16(Kb, DHEAD, t0 + 16,  32), sc1);

    float mt[8];
#pragma unroll
    for (int r = 0; r < 8; ++r) {
      const float a = sc0[r] * SCALEF, b = sc1[r] * SCALEF;
      sc0[r] = a; sc1[r] = b;
      mt[r] = fmaxf(a, b);
    }
    // row-max across the 16 lanes holding one row (C-layout: M = rg*8+r)
#pragma unroll
    for (int off = 1; off < 16; off <<= 1)
#pragma unroll
      for (int r = 0; r < 8; ++r)
        mt[r] = fmaxf(mt[r], __shfl_xor(mt[r], off, 16));

    float ps[8];
#pragma unroll
    for (int r = 0; r < 8; ++r) {
      const float mn = fmaxf(m[r], mt[r]);
      const float alpha = __expf(m[r] - mn);
      m[r] = mn;
      const float p0 = __expf(sc0[r] - mn);
      const float p1 = __expf(sc1[r] - mn);
      lds[(rg * 8 + r) * 32 + rl]      = (_Float16)p0;   // C-layout -> LDS
      lds[(rg * 8 + r) * 32 + 16 + rl] = (_Float16)p1;
      ps[r] = p0 + p1;
      acc0[r] *= alpha; acc1[r] *= alpha; acc2[r] *= alpha; acc3[r] *= alpha;
      l[r] *= alpha;
    }
#pragma unroll
    for (int off = 1; off < 16; off <<= 1)
#pragma unroll
      for (int r = 0; r < 8; ++r)
        ps[r] += __shfl_xor(ps[r], off, 16);
#pragma unroll
    for (int r = 0; r < 8; ++r) l[r] += ps[r];

    // ---- reload P in A-fragment layout (same-wave DS ordering, no barrier) ----
    V16U u;
    u.h[0] = *(const v8h*)(lds + rl * 32 + rg * 8);
    u.h[1] = *(const v8h*)(lds + rl * 32 + 16 + rg * 8);
    const v16h ap = u.v;

    // ---- O += P(16x32) * V(32x64): Vt[d][t] is contiguous in t ----
    acc0 = wmma32(ap, load_b_f16(Vb, S_LEN, 0,  t0), acc0);
    acc1 = wmma32(ap, load_b_f16(Vb, S_LEN, 16, t0), acc1);
    acc2 = wmma32(ap, load_b_f16(Vb, S_LEN, 32, t0), acc2);
    acc3 = wmma32(ap, load_b_f16(Vb, S_LEN, 48, t0), acc3);
  }

  const int bb = bh / NHEAD, h = bh % NHEAD;
#pragma unroll
  for (int r = 0; r < 8; ++r) {
    const float inv = 1.0f / l[r];
    const int s = s0 + rg * 8 + r;
    const size_t base = ((size_t)(s * BATCH + bb)) * DMODEL + h * DHEAD;
    Oh[base + rl]      = (_Float16)(acc0[r] * inv);
    Oh[base + 16 + rl] = (_Float16)(acc1[r] * inv);
    Oh[base + 32 + rl] = (_Float16)(acc2[r] * inv);
    Oh[base + 48 + rl] = (_Float16)(acc3[r] * inv);
  }
}

// ---------------------------------------------------------------------------
// Output projection: out(f32) = Oh[4096x1024] @ Wo^T + bo. 32x32 blocking.
__global__ void out_gemm(const _Float16* __restrict__ X,
                         const _Float16* __restrict__ W,
                         const float* __restrict__ bias,
                         float* __restrict__ out) {
  const int wave = blockIdx.x * (blockDim.x >> 5) + (threadIdx.x >> 5);
  const int tn = wave % (DMODEL / 32);
  const int tm = wave / (DMODEL / 32);
  const int lane = threadIdx.x & 31;
  const int rl = lane & 15, rg = lane >> 4;
  const int m0 = tm * 32, n0 = tn * 32;

  v8f c00 = {}, c01 = {}, c10 = {}, c11 = {};
  for (int k0 = 0; k0 < DMODEL; k0 += 32) {
    const v16h a0 = load_a_f16(X, DMODEL, m0,      k0);
    const v16h a1 = load_a_f16(X, DMODEL, m0 + 16, k0);
    const v16h b0 = load_b_f16(W, DMODEL, n0,      k0);
    const v16h b1 = load_b_f16(W, DMODEL, n0 + 16, k0);
    c00 = wmma32(a0, b0, c00);
    c01 = wmma32(a0, b1, c01);
    c10 = wmma32(a1, b0, c10);
    c11 = wmma32(a1, b1, c11);
  }

  const int o0 = n0 + rl, o1 = o0 + 16;
  const float bi0 = bias[o0], bi1 = bias[o1];
#pragma unroll
  for (int r = 0; r < 8; ++r) {
    const int na = m0 + rg * 8 + r;
    const int nb = na + 16;
    out[(size_t)na * DMODEL + o0] = c00[r] + bi0;
    out[(size_t)na * DMODEL + o1] = c01[r] + bi1;
    out[(size_t)nb * DMODEL + o0] = c10[r] + bi0;
    out[(size_t)nb * DMODEL + o1] = c11[r] + bi1;
  }
}

// ---------------------------------------------------------------------------
extern "C" void kernel_launch(void* const* d_in, const int* in_sizes, int n_in,
                              void* d_out, int out_size, void* d_ws, size_t ws_size,
                              hipStream_t stream) {
  (void)in_sizes; (void)n_in; (void)out_size; (void)ws_size;
  const float* x  = (const float*)d_in[0];
  const float* Wq = (const float*)d_in[1];
  const float* bq = (const float*)d_in[2];
  const float* Wk = (const float*)d_in[3];
  const float* bk = (const float*)d_in[4];
  const float* Wv = (const float*)d_in[5];
  const float* bv = (const float*)d_in[6];
  const float* Wo = (const float*)d_in[7];
  const float* bo = (const float*)d_in[8];
  float* out = (float*)d_out;

  // Workspace partition (halves). Total ~= 50.3 MB.
  _Float16* xh  = (_Float16*)d_ws;                       // 4096*1024
  _Float16* Wqh = xh  + (size_t)NROWS * DMODEL;          // 1024*1024
  _Float16* Wkh = Wqh + (size_t)DMODEL * DMODEL;
  _Float16* Wvh = Wkh + (size_t)DMODEL * DMODEL;
  _Float16* Woh = Wvh + (size_t)DMODEL * DMODEL;
  _Float16* Qh  = Woh + (size_t)DMODEL * DMODEL;         // [b][h][S][64]
  _Float16* Kh  = Qh  + (size_t)NROWS * DMODEL;          // [b][h][S][64]
  _Float16* Vt  = Kh  + (size_t)NROWS * DMODEL;          // [b][h][64][S]
  _Float16* Oh  = Vt  + (size_t)NROWS * DMODEL;          // [n][1024]

  // Stage 0: casts
  cast_f32_to_f16<<<2048, 256, 0, stream>>>(x,  xh,  NROWS * DMODEL);
  cast_f32_to_f16<<<1024, 256, 0, stream>>>(Wq, Wqh, DMODEL * DMODEL);
  cast_f32_to_f16<<<1024, 256, 0, stream>>>(Wk, Wkh, DMODEL * DMODEL);
  cast_f32_to_f16<<<1024, 256, 0, stream>>>(Wv, Wvh, DMODEL * DMODEL);
  cast_f32_to_f16<<<1024, 256, 0, stream>>>(Wo, Woh, DMODEL * DMODEL);

  // Stage 1: QKV projections. 4096 32x32 tiles, 8 waves/block -> 512 blocks.
  qkv_gemm<<<512, 256, 0, stream>>>(xh, Wqh, bq, Qh, 0);
  qkv_gemm<<<512, 256, 0, stream>>>(xh, Wkh, bk, Kh, 1);
  qkv_gemm<<<512, 256, 0, stream>>>(xh, Wvh, bv, Vt, 2);

  // Stage 2: attention. 32 heads * 128 s-tiles = 4096 waves, 4 waves/block.
  flash_attn<<<1024, 128, 0, stream>>>(Qh, Kh, Vt, Oh);

  // Stage 3: output projection into d_out.
  out_gemm<<<512, 256, 0, stream>>>(Oh, Woh, bo, out);
}
